// SSNModel_10368051053194
// MI455X (gfx1250) — matrix operand order repaired and checked by reading, not compiled, not run on hardware
//
#include <hip/hip_runtime.h>
#include <hip/hip_bf16.h>

// ---------------------------------------------------------------------------
// Types / helpers
// ---------------------------------------------------------------------------
typedef __bf16 v16bf __attribute__((ext_vector_type(16)));
typedef float  v8f   __attribute__((ext_vector_type(8)));
typedef int    i32x4 __attribute__((ext_vector_type(4)));

union Frag { v16bf v; uint4 q[2]; unsigned short u[16]; };

#define DEV __device__ __forceinline__

DEV unsigned short f2bf(float f) {            // fp32 -> bf16 (RNE)
    unsigned int u = __float_as_uint(f);
    u += 0x7FFFu + ((u >> 16) & 1u);
    return (unsigned short)(u >> 16);
}
DEV float bf2f(unsigned short h) { return __uint_as_float(((unsigned int)h) << 16); }

static constexpr int   Himg = 384, Wimg = 384, HWimg = Himg * Wimg;
static constexpr int   NW = 32, S = 1024;                // superpixel grid 32x32
static constexpr int   CELL = 12;                        // 384/32
static constexpr int   FDIM = 20;
static constexpr float BN_EPS = 1e-5f;

// ---- optional gfx1250 async global->LDS path (guarded; plain-copy fallback) ----
#if defined(__has_builtin)
#  if __has_builtin(__builtin_amdgcn_global_load_async_to_lds_b128)
#    define HAVE_ASYNC_LDS 1
#  endif
#endif
#ifndef HAVE_ASYNC_LDS
#  define HAVE_ASYNC_LDS 0
#endif

#if HAVE_ASYNC_LDS
typedef __attribute__((address_space(1))) i32x4 gl_i32x4;   // global (AS1)
typedef __attribute__((address_space(3))) i32x4 ld_i32x4;   // LDS    (AS3)
DEV void async_copy16(const unsigned short* gp, unsigned short* lp) {
    __builtin_amdgcn_global_load_async_to_lds_b128(
        (gl_i32x4*)gp, (ld_i32x4*)lp, 0, 0);
}
DEV void wait_async_zero() {
#  if __has_builtin(__builtin_amdgcn_s_wait_asynccnt)
    __builtin_amdgcn_s_wait_asynccnt(0);
#  else
    asm volatile("s_wait_asynccnt 0x0" ::: "memory");
#  endif
}
#endif

// ---------------------------------------------------------------------------
// BN fold:  scale = g/sqrt(v+eps), shift = b - m*scale
// ---------------------------------------------------------------------------
__global__ void bn_fold_kernel(const float* g, const float* b, const float* m,
                               const float* v, float* scale, float* shift) {
    int i = threadIdx.x;
    if (i < 64) {
        float sc = g[i] * __frsqrt_rn(v[i] + BN_EPS);
        scale[i] = sc;
        shift[i] = b[i] - m[i] * sc;
    }
}
__global__ void out_bias_kernel(const float* ob, float* scale, float* shift) {
    int i = threadIdx.x;
    if (i < 64) { scale[i] = 1.0f; shift[i] = (i < 15) ? ob[i] : 0.0f; }
}

// ---------------------------------------------------------------------------
// Pack weights (OIHW f32) -> WMMA B-fragments (bf16), BN scale folded in.
// B[K x 16] per (ntile,kchunk); lane<16 holds K=0..7,16..23 of col n, lane>=16 holds
// K=8..15,24..31 of col n-16; 16 bf16 per lane.  k=(tap*Cpad + cin)
// ---------------------------------------------------------------------------
__global__ void pack_weights_kernel(const float* __restrict__ w,
                                    const float* __restrict__ scale,
                                    unsigned short* __restrict__ bpack,
                                    int Cout, int Cin, int Cpad, int Ktot,
                                    int nkc, int ntiles) {
    int t = blockIdx.x * blockDim.x + threadIdx.x;
    if (t >= ntiles * nkc * 512) return;
    int j    = t & 15;
    int lane = (t >> 4) & 31;
    int rest = t >> 9;
    int kc   = rest % nkc;
    int nt   = rest / nkc;
    int o = nt * 16 + (lane & 15);
    int k = kc * 32 + ((lane < 16) ? 0 : 8) + (j & 7) + ((j >> 3) << 4);
    float val = 0.0f;
    if (o < Cout && k < Ktot) {
        int tap = k / Cpad, c = k - tap * Cpad;
        if (c < Cin) val = w[(o * Cin + c) * 9 + tap] * scale[o];
    }
    bpack[t] = f2bf(val);
}

// ---------------------------------------------------------------------------
// Pack input x: f32 CHW [5,H,W] -> bf16 HWC, Cpad=8
// ---------------------------------------------------------------------------
__global__ void pack_input_kernel(const float* __restrict__ x,
                                  unsigned short* __restrict__ dst) {
    int t = blockIdx.x * blockDim.x + threadIdx.x;
    if (t >= HWimg * 8) return;
    int p = t >> 3, c = t & 7;
    dst[t] = f2bf((c < 5) ? x[c * HWimg + p] : 0.0f);
}

// ---------------------------------------------------------------------------
// Implicit-GEMM conv3x3 (pad 1), v_wmma_f32_16x16x32_bf16.
// Block: 32 pixels x (NTILES*16) outputs.  Wave w owns output tile w and TWO
// M-tiles (independent accumulator chains sharing one B fragment).
// A is built from a row-contiguous 3x34xCpad halo staged in LDS.
// ---------------------------------------------------------------------------
template <int Cpad, int Ktot, int Kpad, int NKC, int NTILES, int COUT, bool F32OUT>
__global__ void conv_wmma_t(const unsigned short* __restrict__ src,
                            const unsigned short* __restrict__ bpack,
                            const float* __restrict__ bias,
                            unsigned short* __restrict__ dstBf,
                            float* __restrict__ dstF,
                            int W, int H, int outCpad) {
    constexpr int C8 = Cpad / 8;
    __shared__ __align__(16) unsigned short halo[3 * 34 * Cpad];

    const int tid = threadIdx.x;
    const int p0  = blockIdx.x * 32;           // 32 consecutive pixels, W % 32 == 0
    const int y0  = p0 / W, x0 = p0 - y0 * W;

    // ---- stage halo rows (row-contiguous 16B chunks) ----
    for (int g = tid; g < 3 * 34 * C8; g += blockDim.x) {
        int r   = g / (34 * C8);
        int rem = g - r * (34 * C8);
        int px  = rem / C8;
        int c8  = rem - px * C8;
        int sy = y0 - 1 + r, sx = x0 - 1 + px;
        unsigned short* lp = halo + (r * 34 + px) * Cpad + c8 * 8;
        if (sy >= 0 && sy < H && sx >= 0 && sx < W) {
            const unsigned short* gp = src + ((size_t)(sy * W + sx)) * Cpad + c8 * 8;
#if HAVE_ASYNC_LDS
            async_copy16(gp, lp);
#else
            *(uint4*)lp = *(const uint4*)gp;
#endif
        } else {
            uint4 z = {0u, 0u, 0u, 0u};
            *(uint4*)lp = z;
        }
    }
#if HAVE_ASYNC_LDS
    wait_async_zero();
#endif
    __syncthreads();

    const int wave = tid >> 5;
    if (wave < NTILES) {
        const int lane  = tid & 31;
        const int mA    = lane & 15;
        const int khalf = (lane >= 16) ? 8 : 0;

        v8f acc0 = {0.f,0.f,0.f,0.f,0.f,0.f,0.f,0.f};
        v8f acc1 = {0.f,0.f,0.f,0.f,0.f,0.f,0.f,0.f};
        const uint4 z4 = {0u, 0u, 0u, 0u};

#pragma unroll
        for (int kc = 0; kc < NKC; ++kc) {
            // two 8-element K-runs: [k0,k0+8) and [k0+16,k0+24)
            const int k0 = kc * 32 + khalf;
            const int k1 = k0 + 16;
            // tap = (dy+1)*3 + (dx+1); halo idx = (rowSel*34 + m + dxSel)*Cpad + c
            const int tap0 = k0 / Cpad, c0 = k0 - tap0 * Cpad;
            const int tap1 = k1 / Cpad, c1 = k1 - tap1 * Cpad;
            const int ry0 = tap0 / 3, rx0 = tap0 - ry0 * 3;
            const int ry1 = tap1 / 3, rx1 = tap1 - ry1 * 3;
            const bool v0 = (k0 < Ktot), v1 = (k1 < Ktot);

            Frag a0, a1, b;
            {
                const unsigned short* h00 = halo + (ry0 * 34 + mA + rx0) * Cpad + c0;
                const unsigned short* h01 = halo + (ry1 * 34 + mA + rx1) * Cpad + c1;
                a0.q[0] = v0 ? *(const uint4*)h00 : z4;
                a0.q[1] = v1 ? *(const uint4*)h01 : z4;
                const unsigned short* h10 = h00 + 16 * Cpad;   // second M tile (+16 px)
                const unsigned short* h11 = h01 + 16 * Cpad;
                a1.q[0] = v0 ? *(const uint4*)h10 : z4;
                a1.q[1] = v1 ? *(const uint4*)h11 : z4;
            }
            const uint4* bp = (const uint4*)(bpack + ((size_t)(wave * NKC + kc) * 32 + lane) * 16);
            b.q[0] = bp[0];
            b.q[1] = bp[1];
            if (kc + 1 < NKC)
                __builtin_prefetch(bpack + ((size_t)(wave * NKC + kc + 1) * 32 + lane) * 16, 0, 1);

            acc0 = __builtin_amdgcn_wmma_f32_16x16x32_bf16(
                       false, a0.v, false, b.v, (short)0, acc0, false, false);
            acc1 = __builtin_amdgcn_wmma_f32_16x16x32_bf16(
                       false, a1.v, false, b.v, (short)0, acc1, false, false);
        }

        // ---- epilogue: bias + ReLU.  D: lane = (col n, half), VGPR r = row ----
        const int n     = lane & 15;
        const int mbase = (lane >= 16) ? 8 : 0;
        const int ch    = wave * 16 + n;
        if (ch < COUT) {
            const float bv = bias[ch];
#pragma unroll
            for (int r = 0; r < 8; ++r) {
                int pix0 = p0 + mbase + r;
                float f0 = fmaxf(acc0[r] + bv, 0.0f);
                float f1 = fmaxf(acc1[r] + bv, 0.0f);
                if (F32OUT) {
                    dstF[(size_t)pix0 * outCpad + ch]        = f0;
                    dstF[(size_t)(pix0 + 16) * outCpad + ch] = f1;
                } else {
                    dstBf[(size_t)pix0 * outCpad + ch]        = f2bf(f0);
                    dstBf[(size_t)(pix0 + 16) * outCpad + ch] = f2bf(f1);
                }
            }
        }
    }
}

// ---------------------------------------------------------------------------
// MaxPool 3x3 stride 2 pad 1 on bf16 HWC (C = 64)
// ---------------------------------------------------------------------------
__global__ void maxpool_kernel(const unsigned short* __restrict__ in,
                               unsigned short* __restrict__ out,
                               int Hi, int Wi, int Ho, int Wo) {
    int t = blockIdx.x * blockDim.x + threadIdx.x;
    if (t >= Ho * Wo * 64) return;
    int c = t & 63, q = t >> 6;
    int xo = q % Wo, yo = q / Wo;
    float m = -INFINITY;
    for (int ky = 0; ky < 3; ++ky)
        for (int kx = 0; kx < 3; ++kx) {
            int yi = 2 * yo - 1 + ky, xi = 2 * xo - 1 + kx;
            if (yi >= 0 && yi < Hi && xi >= 0 && xi < Wi)
                m = fmaxf(m, bf2f(in[(yi * Wi + xi) * 64 + c]));
        }
    out[t] = f2bf(m);
}

// ---------------------------------------------------------------------------
// Concat [HW x 200] bf16: x(5) | s1(64) | up2(s2,64) | up4(s3,64) | pad(3)
// Bilinear, half-pixel centers (align_corners=False)
// ---------------------------------------------------------------------------
DEV void bil_setup(int d, float scale, int Hs, int& i0, int& i1, float& w1) {
    float s = ((float)d + 0.5f) * scale - 0.5f;
    float f = floorf(s);
    w1 = s - f;
    i0 = (int)f; i1 = i0 + 1;
    i0 = min(max(i0, 0), Hs - 1);
    i1 = min(max(i1, 0), Hs - 1);
}
__global__ void build_cat_kernel(const float* __restrict__ x,
                                 const unsigned short* __restrict__ s1,
                                 const unsigned short* __restrict__ s2,
                                 const unsigned short* __restrict__ s3,
                                 unsigned short* __restrict__ cat) {
    int p = blockIdx.x * blockDim.x + threadIdx.x;
    if (p >= HWimg) return;
    int y = p / Wimg, xx = p - y * Wimg;
    unsigned short* row = cat + (size_t)p * 200;
    for (int c = 0; c < 5; ++c)  row[c]     = f2bf(x[c * HWimg + p]);
    for (int c = 0; c < 64; ++c) row[5 + c] = s1[(size_t)p * 64 + c];
    int y0, y1, x0, x1; float wy, wx;
    bil_setup(y, 0.5f, 192, y0, y1, wy);
    bil_setup(xx, 0.5f, 192, x0, x1, wx);
    for (int c = 0; c < 64; ++c) {
        float v00 = bf2f(s2[(y0 * 192 + x0) * 64 + c]);
        float v01 = bf2f(s2[(y0 * 192 + x1) * 64 + c]);
        float v10 = bf2f(s2[(y1 * 192 + x0) * 64 + c]);
        float v11 = bf2f(s2[(y1 * 192 + x1) * 64 + c]);
        row[69 + c] = f2bf((1 - wy) * ((1 - wx) * v00 + wx * v01) +
                           wy * ((1 - wx) * v10 + wx * v11));
    }
    bil_setup(y, 0.25f, 96, y0, y1, wy);
    bil_setup(xx, 0.25f, 96, x0, x1, wx);
    for (int c = 0; c < 64; ++c) {
        float v00 = bf2f(s3[(y0 * 96 + x0) * 64 + c]);
        float v01 = bf2f(s3[(y0 * 96 + x1) * 64 + c]);
        float v10 = bf2f(s3[(y1 * 96 + x0) * 64 + c]);
        float v11 = bf2f(s3[(y1 * 96 + x1) * 64 + c]);
        row[133 + c] = f2bf((1 - wy) * ((1 - wx) * v00 + wx * v01) +
                            wy * ((1 - wx) * v10 + wx * v11));
    }
    row[197] = 0; row[198] = 0; row[199] = 0;
}

__global__ void pf_x_kernel(const float* __restrict__ x, float* __restrict__ pf) {
    int t = blockIdx.x * blockDim.x + threadIdx.x;
    if (t >= HWimg * 5) return;
    int p = t / 5, c = t - p * 5;
    pf[(size_t)p * FDIM + 15 + c] = x[c * HWimg + p];
}

// ---------------------------------------------------------------------------
// SSN
// ---------------------------------------------------------------------------
__global__ void sf_init_kernel(const float* __restrict__ pf, float* __restrict__ sf) {
    int t = blockIdx.x * blockDim.x + threadIdx.x;
    if (t >= S * FDIM) return;
    int s = t / FDIM, c = t - s * FDIM;
    int cy = s >> 5, cx = s & 31;
    float acc = 0.0f;
    for (int iy = 0; iy < CELL; ++iy)
        for (int ix = 0; ix < CELL; ++ix)
            acc += pf[(size_t)((cy * CELL + iy) * Wimg + cx * CELL + ix) * FDIM + c];
    sf[t] = acc * (1.0f / (CELL * CELL));
}

__global__ void zero_numden_kernel(float* __restrict__ num, float* __restrict__ den) {
    int t = blockIdx.x * blockDim.x + threadIdx.x;
    if (t < S * FDIM) num[t] = 0.0f;
    else if (t < S * FDIM + S) den[t - S * FDIM] = 0.0f;
}

__constant__ int c_offs[9] = {-33, -32, -31, -1, 0, 1, 31, 32, 33};

__global__ void ssn_iter_kernel(const float* __restrict__ pf,
                                const float* __restrict__ sf,
                                float* __restrict__ num, float* __restrict__ den,
                                int* __restrict__ rel) {
    __shared__ float nbsf[9][FDIM];
    __shared__ float accum[9][FDIM + 1];
    const int s = blockIdx.x, t = threadIdx.x;

    if (t < 9 * FDIM) {
        int o = t / FDIM, c = t - o * FDIM;
        int idx = min(max(s + c_offs[o], 0), S - 1);
        nbsf[o][c] = sf[idx * FDIM + c];
    }
    if (t < 9 * (FDIM + 1)) accum[t / (FDIM + 1)][t % (FDIM + 1)] = 0.0f;
    __syncthreads();

    if (t < CELL * CELL) {
        int cy = s >> 5, cx = s & 31;
        int ly = t / CELL, lx = t - ly * CELL;
        int p = (cy * CELL + ly) * Wimg + cx * CELL + lx;
        float pfl[FDIM];
#pragma unroll
        for (int c = 0; c < FDIM; ++c) pfl[c] = pf[(size_t)p * FDIM + c];

        float dist[9]; float mind = 3.4e38f; int best = 0;
#pragma unroll
        for (int o = 0; o < 9; ++o) {
            float d = 0.0f;
#pragma unroll
            for (int c = 0; c < FDIM; ++c) {
                float df = pfl[c] - nbsf[o][c];
                d += df * df;
            }
            dist[o] = d;
            if (d < mind) { mind = d; best = o; }
        }
        rel[p] = best;

        float e[9], esum = 0.0f;
#pragma unroll
        for (int o = 0; o < 9; ++o) { e[o] = __expf(mind - dist[o]); esum += e[o]; }
        float inv = 1.0f / esum;
#pragma unroll
        for (int o = 0; o < 9; ++o) {
            int raw = s + c_offs[o];
            float w = (raw >= 0 && raw < S) ? e[o] * inv : 0.0f;
#pragma unroll
            for (int c = 0; c < FDIM; ++c) atomicAdd(&accum[o][c], w * pfl[c]);
            atomicAdd(&accum[o][FDIM], w);
        }
    }
    __syncthreads();

    if (t < 9 * (FDIM + 1)) {
        int o = t / (FDIM + 1), j = t - o * (FDIM + 1);
        int raw = s + c_offs[o];
        int idx = (raw >= 0 && raw < S) ? raw : 0;   // masked -> seg 0 with weight 0
        float v = accum[o][j];
        if (j < FDIM) atomicAdd(&num[idx * FDIM + j], v);
        else          atomicAdd(&den[idx], v);
    }
}

__global__ void sf_update_kernel(const float* __restrict__ num,
                                 const float* __restrict__ den,
                                 float* __restrict__ sf) {
    int t = blockIdx.x * blockDim.x + threadIdx.x;
    if (t >= S * FDIM) return;
    sf[t] = num[t] / (den[t / FDIM] + 1e-16f);
}

__global__ void labels_kernel(const int* __restrict__ rel, int* __restrict__ out) {
    int p = blockIdx.x * blockDim.x + threadIdx.x;
    if (p >= HWimg) return;
    int y = p / Wimg, x = p - y * Wimg;
    int s = (y / CELL) * NW + (x / CELL);
    out[p] = s + c_offs[rel[p]];
}

// ---------------------------------------------------------------------------
// Host side
// ---------------------------------------------------------------------------
static inline int ceil_div(int a, int b) { return (a + b - 1) / b; }

extern "C" void kernel_launch(void* const* d_in, const int* in_sizes, int n_in,
                              void* d_out, int out_size, void* d_ws, size_t ws_size,
                              hipStream_t stream) {
    (void)in_sizes; (void)n_in; (void)out_size; (void)ws_size;
    const float* x = (const float*)d_in[0];
    const float* W[7]; const float* G[6]; const float* B[6]; const float* M[6]; const float* V[6];
    for (int l = 0; l < 6; ++l) {
        W[l] = (const float*)d_in[1 + 5 * l + 0];
        G[l] = (const float*)d_in[1 + 5 * l + 1];
        B[l] = (const float*)d_in[1 + 5 * l + 2];
        M[l] = (const float*)d_in[1 + 5 * l + 3];
        V[l] = (const float*)d_in[1 + 5 * l + 4];
    }
    W[6] = (const float*)d_in[31];
    const float* out_b = (const float*)d_in[32];

    // ---- workspace carve-out ----
    char* ws = (char*)d_ws;
    size_t cur = 0;
    auto carve = [&](size_t bytes) -> char* {
        cur = (cur + 255) & ~(size_t)255;
        char* p = ws + cur; cur += bytes; return p;
    };
    float* scaleA = (float*)carve(7 * 64 * sizeof(float));
    float* shiftA = (float*)carve(7 * 64 * sizeof(float));
    unsigned short* xbf   = (unsigned short*)carve((size_t)HWimg * 8 * 2);
    unsigned short* s1tmp = (unsigned short*)carve((size_t)HWimg * 64 * 2);
    unsigned short* s1    = (unsigned short*)carve((size_t)HWimg * 64 * 2);
    unsigned short* pool1 = (unsigned short*)carve((size_t)192 * 192 * 64 * 2);
    unsigned short* s2tmp = (unsigned short*)carve((size_t)192 * 192 * 64 * 2);
    unsigned short* s2    = (unsigned short*)carve((size_t)192 * 192 * 64 * 2);
    unsigned short* pool2 = (unsigned short*)carve((size_t)96 * 96 * 64 * 2);
    unsigned short* s3tmp = (unsigned short*)carve((size_t)96 * 96 * 64 * 2);
    unsigned short* s3    = (unsigned short*)carve((size_t)96 * 96 * 64 * 2);
    unsigned short* cat   = (unsigned short*)carve((size_t)HWimg * 200 * 2);
    float* pf  = (float*)carve((size_t)HWimg * FDIM * sizeof(float));
    float* sf  = (float*)carve((size_t)S * FDIM * sizeof(float));
    float* num = (float*)carve((size_t)S * FDIM * sizeof(float));
    float* den = (float*)carve((size_t)S * sizeof(float));
    int*   rel = (int*)carve((size_t)HWimg * sizeof(int));

    struct LG { int Cin, Cpad, Ktot, nkc, ntiles, Cout; };
    const LG lg[7] = {
        {  5,   8,   72,  3, 4, 64},
        { 64,  64,  576, 18, 4, 64},
        { 64,  64,  576, 18, 4, 64},
        { 64,  64,  576, 18, 4, 64},
        { 64,  64,  576, 18, 4, 64},
        { 64,  64,  576, 18, 4, 64},
        {197, 200, 1800, 57, 1, 15},
    };
    unsigned short* bpk[7];
    for (int l = 0; l < 7; ++l)
        bpk[l] = (unsigned short*)carve((size_t)lg[l].ntiles * lg[l].nkc * 512 * 2);

    // ---- BN fold + weight packing ----
    for (int l = 0; l < 6; ++l)
        bn_fold_kernel<<<1, 64, 0, stream>>>(G[l], B[l], M[l], V[l],
                                             scaleA + l * 64, shiftA + l * 64);
    out_bias_kernel<<<1, 64, 0, stream>>>(out_b, scaleA + 6 * 64, shiftA + 6 * 64);
    for (int l = 0; l < 7; ++l) {
        int total = lg[l].ntiles * lg[l].nkc * 512;
        pack_weights_kernel<<<ceil_div(total, 256), 256, 0, stream>>>(
            W[l], scaleA + l * 64, bpk[l],
            lg[l].Cout, lg[l].Cin, lg[l].Cpad, lg[l].Ktot, lg[l].nkc, lg[l].ntiles);
    }
    pack_input_kernel<<<ceil_div(HWimg * 8, 256), 256, 0, stream>>>(x, xbf);

    // ---- feature extractor (templated WMMA convs) ----
    conv_wmma_t<8, 72, 96, 3, 4, 64, false>
        <<<HWimg / 32, 128, 0, stream>>>(xbf, bpk[0], shiftA + 0 * 64,
                                         s1tmp, nullptr, 384, 384, 64);
    conv_wmma_t<64, 576, 576, 18, 4, 64, false>
        <<<HWimg / 32, 128, 0, stream>>>(s1tmp, bpk[1], shiftA + 1 * 64,
                                         s1, nullptr, 384, 384, 64);
    maxpool_kernel<<<ceil_div(192 * 192 * 64, 256), 256, 0, stream>>>(s1, pool1, 384, 384, 192, 192);
    conv_wmma_t<64, 576, 576, 18, 4, 64, false>
        <<<(192 * 192) / 32, 128, 0, stream>>>(pool1, bpk[2], shiftA + 2 * 64,
                                               s2tmp, nullptr, 192, 192, 64);
    conv_wmma_t<64, 576, 576, 18, 4, 64, false>
        <<<(192 * 192) / 32, 128, 0, stream>>>(s2tmp, bpk[3], shiftA + 3 * 64,
                                               s2, nullptr, 192, 192, 64);
    maxpool_kernel<<<ceil_div(96 * 96 * 64, 256), 256, 0, stream>>>(s2, pool2, 192, 192, 96, 96);
    conv_wmma_t<64, 576, 576, 18, 4, 64, false>
        <<<(96 * 96) / 32, 128, 0, stream>>>(pool2, bpk[4], shiftA + 4 * 64,
                                             s3tmp, nullptr, 96, 96, 64);
    conv_wmma_t<64, 576, 576, 18, 4, 64, false>
        <<<(96 * 96) / 32, 128, 0, stream>>>(s3tmp, bpk[5], shiftA + 5 * 64,
                                             s3, nullptr, 96, 96, 64);
    build_cat_kernel<<<ceil_div(HWimg, 256), 256, 0, stream>>>(x, s1, s2, s3, cat);
    conv_wmma_t<200, 1800, 1824, 57, 1, 15, true>
        <<<HWimg / 32, 64, 0, stream>>>(cat, bpk[6], shiftA + 6 * 64,
                                        nullptr, pf, 384, 384, FDIM);
    pf_x_kernel<<<ceil_div(HWimg * 5, 256), 256, 0, stream>>>(x, pf);

    // ---- SSN iterations ----
    sf_init_kernel<<<ceil_div(S * FDIM, 256), 256, 0, stream>>>(pf, sf);
    for (int it = 0; it < 10; ++it) {
        zero_numden_kernel<<<ceil_div(S * (FDIM + 1), 256), 256, 0, stream>>>(num, den);
        ssn_iter_kernel<<<S, 192, 0, stream>>>(pf, sf, num, den, rel);
        sf_update_kernel<<<ceil_div(S * FDIM, 256), 256, 0, stream>>>(num, den, sf);
    }
    labels_kernel<<<ceil_div(HWimg, 256), 256, 0, stream>>>(rel, (int*)d_out);
}
